// NeuralNetwork_72507637891261
// MI455X (gfx1250) — matrix operand order repaired and checked by reading, not compiled, not run on hardware
//
#include <hip/hip_runtime.h>
#include <hip/hip_bf16.h>

// ---------------------------------------------------------------------------
// 2-layer LSTM (T=1024, B=64, Din=256, H=512) + output projection for gfx1250.
// Strategy: bf16 WMMA (v_wmma_f32_16x16x32_bf16) everywhere; input projections
// precomputed off the serial path; persistent recurrent kernel with W_hh
// pre-swizzled into LDS in WMMA-B fragment layout and a per-timestep global
// grid barrier. fp32 accumulation throughout; gate math in fp32.
// Workspace required: ~420 MB.
// ---------------------------------------------------------------------------

typedef __bf16 bh;
typedef __attribute__((ext_vector_type(16))) __bf16 v16bf;
typedef __attribute__((ext_vector_type(8)))  __bf16 v8bf;
typedef __attribute__((ext_vector_type(8)))  float  v8f;

#define WMMA_BF16(a, b, c) \
  __builtin_amdgcn_wmma_f32_16x16x32_bf16(false, (a), false, (b), (short)0, (c), false, false)

static constexpr int T_    = 1024;
static constexpr int B_    = 64;
static constexpr int H_    = 512;
static constexpr int G4H   = 2048;      // 4*H
static constexpr int NWG   = 32;        // workgroups in recurrent kernel
static constexpr int COLS  = H_ / NWG;  // 16 h-columns per WG
static constexpr int NT    = 4;         // gate n-tiles per WG (one per gate)
static constexpr int KB    = H_ / 32;   // 16 k-blocks of 32

// A-fragment in this code's contiguous k order (B matches): lanes 0-15 hold K 0-15,
// lanes 16-31 hold K 16-31, row = lane&15.  Two 16B loads + shuffle.
__device__ inline v16bf load_a_frag(const bh* p /* row base + kblock*32 */, int lane) {
  const int base = (lane & 16) ? 16 : 0;
  v8bf lo = *(const v8bf*)(p + base);
  v8bf hi = *(const v8bf*)(p + base + 8);
  return __builtin_shufflevector(lo, hi, 0,1,2,3,4,5,6,7,8,9,10,11,12,13,14,15);
}

__device__ inline float sigm(float x) { return 1.0f / (1.0f + __expf(-x)); }

// ---------------------------------------------------------------------------
// Generic bf16 GEMM with bias:  Out[M,N] = A[M,K] * Bm[N,K]^T + bias0 (+bias1)
// Block tile 128(M) x 128(N), 8 waves: wave w owns rows w*16..w*16+15,
// all 8 n-tiles (acc = 8 x v8f).  All dims multiples of 128/32 -> no guards.
// ---------------------------------------------------------------------------
template <bool OUT_BF16>
__global__ void __launch_bounds__(256)
gemm_bias_kernel(const bh* __restrict__ A, const bh* __restrict__ Bm,
                 const float* __restrict__ bias0, const float* __restrict__ bias1,
                 void* __restrict__ Out, int N, int K) {
  const int lane    = threadIdx.x & 31;
  const int wave    = threadIdx.x >> 5;
  const int lanelow = lane & 15;
  const int koff    = (lane & 16) ? 16 : 0;
  const int mrow0   = blockIdx.x * 128 + wave * 16;
  const int ncol0   = blockIdx.y * 128;

  v8f acc[8] = {};
  const bh* arow = A + (size_t)(mrow0 + lanelow) * K;
  for (int kb = 0; kb < K; kb += 32) {
    v16bf a = load_a_frag(arow + kb, lane);
#pragma unroll
    for (int nt = 0; nt < 8; ++nt) {
      // B-fragment (32x16): lane&15 = column, 16 contiguous K at koff.
      const bh* bp = Bm + (size_t)(ncol0 + nt * 16 + lanelow) * K + kb + koff;
      v16bf b = *(const v16bf*)bp;
      acc[nt] = WMMA_BF16(a, b, acc[nt]);
    }
  }
  // C/D layout: vgpr r -> row r (+8 for lanes 16-31), col = lane&15.
  const int rb = (lane & 16) ? 8 : 0;
#pragma unroll
  for (int nt = 0; nt < 8; ++nt) {
    const int col = ncol0 + nt * 16 + lanelow;
    float bv = bias0[col] + (bias1 ? bias1[col] : 0.0f);
#pragma unroll
    for (int r = 0; r < 8; ++r) {
      const size_t idx = (size_t)(mrow0 + rb + r) * N + col;
      const float v = acc[nt][r] + bv;
      if constexpr (OUT_BF16) ((bh*)Out)[idx] = (bh)v;
      else                    ((float*)Out)[idx] = v;
    }
  }
}

// ---------------------------------------------------------------------------
// Persistent recurrent LSTM layer.  32 WGs; WG wg owns h columns
// [wg*16, wg*16+16) i.e. gate columns g*512 + wg*16 + j for g=0..3.
// Per step: gates = hs[t] @ Whh_slice^T (WMMA, B-operand from LDS) + xproj,
// elementwise LSTM update, write hs[t+1] (bf16), grid barrier.
// ---------------------------------------------------------------------------
__global__ void __launch_bounds__(256)
lstm_rec_kernel(const bh* __restrict__ xproj,      // [T, B, 4H] bf16
                const float* __restrict__ Whh,     // [4H, H] fp32
                bh* __restrict__ hs,               // [(T+1), B, H] bf16 (slice0 zeroed)
                unsigned* __restrict__ cnt,        // [T] zeroed counters
                float* __restrict__ hn, float* __restrict__ cn) {  // [B, H]
  __shared__ __align__(32) bh lds_w[NT * KB * 32 * 16];   // 32768 bh = 64 KB
  __shared__ float gates[B_ * 4 * COLS];                  // 64x64 f32 = 16 KB
  __shared__ float cbuf[B_ * COLS];                       // 64x16 f32 =  4 KB

  const int tid     = threadIdx.x;
  const int lane    = tid & 31;
  const int wave    = tid >> 5;
  const int mtile   = wave & 3;   // 4 M-tiles of 16 batch rows
  const int nhalf   = wave >> 1 & 0; // placeholder (see below)
  (void)nhalf;
  const int npair   = wave >> 2;  // 0/1 -> n-tiles {0,1} / {2,3}
  const int wg      = blockIdx.x;
  const int lanelow = lane & 15;

  // --- preload W_hh slice into LDS in WMMA-B fragment layout (fp32 -> bf16).
  // frag f = (nt, kb, lane): 16 bf16 = Whh[gcol][kb*32 + koff .. +15]
  for (int f = tid; f < NT * KB * 32; f += 256) {
    const int lf = f & 31;
    const int kb = (f >> 5) & (KB - 1);
    const int nt = f >> 9;                       // 0..3 == gate index
    const int gcol = nt * 512 + wg * COLS + (lf & 15);
    const int kk = kb * 32 + ((lf & 16) ? 16 : 0);
    const float* src = Whh + (size_t)gcol * H_ + kk;
    bh* dst = lds_w + (size_t)f * 16;
#pragma unroll
    for (int e = 0; e < 16; ++e) dst[e] = (bh)src[e];
  }
  for (int i = tid; i < B_ * COLS; i += 256) cbuf[i] = 0.0f;
  __syncthreads();

  for (int t = 0; t < T_; ++t) {
    // ---- GEMM: gates(64 x 64) += hs[t](64 x 512) * Wslice^T
    const bh* hsrc = hs + (size_t)t * (B_ * H_);
    const bh* arow = hsrc + (size_t)(mtile * 16 + lanelow) * H_;
    v8f acc[2] = {};
    for (int kb = 0; kb < KB; ++kb) {
      v16bf a = load_a_frag(arow + kb * 32, lane);
#pragma unroll
      for (int j = 0; j < 2; ++j) {
        const int nt = npair * 2 + j;
        v16bf b = *(const v16bf*)(lds_w + (size_t)((nt * KB + kb) * 32 + lane) * 16);
        acc[j] = WMMA_BF16(a, b, acc[j]);
      }
    }
    const int rb = (lane & 16) ? 8 : 0;
#pragma unroll
    for (int j = 0; j < 2; ++j) {
      const int col = (npair * 2 + j) * 16 + lanelow;   // local gate col 0..63
#pragma unroll
      for (int r = 0; r < 8; ++r)
        gates[(mtile * 16 + rb + r) * (4 * COLS) + col] = acc[j][r];
    }
    __syncthreads();

    // ---- elementwise LSTM update for our 64x16 cells
    const bh* xp = xproj + (size_t)t * (B_ * G4H);
    for (int cell = tid; cell < B_ * COLS; cell += 256) {
      const int b = cell >> 4;
      const int j = cell & (COLS - 1);
      const int hcol = wg * COLS + j;
      const float* g = gates + b * (4 * COLS);
      const float gi = g[j]             + (float)xp[(size_t)b * G4H + hcol];
      const float gf = g[COLS + j]      + (float)xp[(size_t)b * G4H + 512 + hcol];
      const float gg = g[2 * COLS + j]  + (float)xp[(size_t)b * G4H + 1024 + hcol];
      const float go = g[3 * COLS + j]  + (float)xp[(size_t)b * G4H + 1536 + hcol];
      const float cold = cbuf[cell];
      const float cnew = sigm(gf) * cold + sigm(gi) * tanhf(gg);
      const float hval = sigm(go) * tanhf(cnew);
      cbuf[cell] = cnew;
      hs[(size_t)(t + 1) * (B_ * H_) + (size_t)b * H_ + hcol] = (bh)hval;
      if (t == T_ - 1) {
        hn[(size_t)b * H_ + hcol] = hval;
        cn[(size_t)b * H_ + hcol] = cnew;
      }
    }

    // ---- grid-wide barrier (32 fully-resident WGs; per-step counter)
    __syncthreads();
    __threadfence();
    if (tid == 0) {
      __hip_atomic_fetch_add(&cnt[t], 1u, __ATOMIC_RELEASE, __HIP_MEMORY_SCOPE_AGENT);
      while (__hip_atomic_load(&cnt[t], __ATOMIC_ACQUIRE, __HIP_MEMORY_SCOPE_AGENT) < (unsigned)NWG)
        __builtin_amdgcn_s_sleep(2);
    }
    __syncthreads();
    __threadfence();
  }
}

// fp32 -> bf16 bulk convert (n multiple of 4)
__global__ void cvt_f32_bf16_kernel(const float* __restrict__ src, bh* __restrict__ dst, int n) {
  const int i4 = (blockIdx.x * blockDim.x + threadIdx.x) * 4;
  if (i4 < n) {
    float4 v = *(const float4*)(src + i4);
    dst[i4 + 0] = (bh)v.x; dst[i4 + 1] = (bh)v.y;
    dst[i4 + 2] = (bh)v.z; dst[i4 + 3] = (bh)v.w;
  }
}

// zero h(0) slices and barrier counters
__global__ void init_state_kernel(bh* __restrict__ h1, bh* __restrict__ h2,
                                  unsigned* __restrict__ cnt) {
  const int i = blockIdx.x * blockDim.x + threadIdx.x;  // 32768 threads
  h1[i] = (bh)0.0f;
  h2[i] = (bh)0.0f;
  if (i < 2 * T_) cnt[i] = 0u;
}

extern "C" void kernel_launch(void* const* d_in, const int* in_sizes, int n_in,
                              void* d_out, int out_size, void* d_ws, size_t ws_size,
                              hipStream_t stream) {
  (void)in_sizes; (void)n_in; (void)out_size; (void)ws_size;
  const float* x     = (const float*)d_in[0];
  const float* W_ih0 = (const float*)d_in[1];
  const float* W_hh0 = (const float*)d_in[2];
  const float* b_ih0 = (const float*)d_in[3];
  const float* b_hh0 = (const float*)d_in[4];
  const float* W_ih1 = (const float*)d_in[5];
  const float* W_hh1 = (const float*)d_in[6];
  const float* b_ih1 = (const float*)d_in[7];
  const float* b_hh1 = (const float*)d_in[8];
  const float* W_out = (const float*)d_in[9];
  const float* b_out = (const float*)d_in[10];

  float* out = (float*)d_out;                        // [T, B, 256]
  float* hn  = out + (size_t)T_ * B_ * 256;          // [2, B, H]
  float* cn  = hn + (size_t)2 * B_ * H_;             // [2, B, H]

  // ---- workspace partition (256B aligned), ~420 MB total
  char* ws = (char*)d_ws;
  size_t off = 0;
  auto take = [&](size_t bytes) -> char* {
    char* p = ws + off;
    off += (bytes + 255) & ~(size_t)255;
    return p;
  };
  const size_t M = (size_t)T_ * B_;                              // 65536
  bh* xbf   = (bh*)take(M * 256 * sizeof(bh));                   // x in bf16
  bh* wih0b = (bh*)take((size_t)G4H * 256 * sizeof(bh));
  bh* wih1b = (bh*)take((size_t)G4H * H_ * sizeof(bh));
  bh* woutb = (bh*)take((size_t)256 * H_ * sizeof(bh));
  bh* xproj = (bh*)take(M * G4H * sizeof(bh));                   // reused per layer
  bh* hs1   = (bh*)take((size_t)(T_ + 1) * B_ * H_ * sizeof(bh));
  bh* hs2   = (bh*)take((size_t)(T_ + 1) * B_ * H_ * sizeof(bh));
  unsigned* cnt = (unsigned*)take((size_t)2 * T_ * sizeof(unsigned));

  // ---- stage 0: conversions + state init
  cvt_f32_bf16_kernel<<<(int)(M * 256 / 4 / 256), 256, 0, stream>>>(x, xbf, (int)(M * 256));
  cvt_f32_bf16_kernel<<<G4H * 256 / 4 / 256, 256, 0, stream>>>(W_ih0, wih0b, G4H * 256);
  cvt_f32_bf16_kernel<<<G4H * H_ / 4 / 256, 256, 0, stream>>>(W_ih1, wih1b, G4H * H_);
  cvt_f32_bf16_kernel<<<256 * H_ / 4 / 256, 256, 0, stream>>>(W_out, woutb, 256 * H_);
  init_state_kernel<<<(B_ * H_) / 256, 256, 0, stream>>>(hs1, hs2, cnt);

  // ---- layer 0: xproj = x @ W_ih0^T + b_ih0 + b_hh0  (M=65536, N=2048, K=256)
  gemm_bias_kernel<true><<<dim3(512, 16), 256, 0, stream>>>(
      xbf, wih0b, b_ih0, b_hh0, xproj, G4H, 256);
  lstm_rec_kernel<<<NWG, 256, 0, stream>>>(xproj, W_hh0, hs1, cnt, hn, cn);

  // ---- layer 1: xproj = h1s @ W_ih1^T + b_ih1 + b_hh1 (K=512)
  gemm_bias_kernel<true><<<dim3(512, 16), 256, 0, stream>>>(
      hs1 + (size_t)B_ * H_, wih1b, b_ih1, b_hh1, xproj, G4H, H_);
  lstm_rec_kernel<<<NWG, 256, 0, stream>>>(xproj, W_hh1, hs2, cnt + T_,
                                           hn + (size_t)B_ * H_, cn + (size_t)B_ * H_);

  // ---- output: out = h2s @ W_out^T + b_out (M=65536, N=256, K=512)
  gemm_bias_kernel<false><<<dim3(512, 2), 256, 0, stream>>>(
      hs2 + (size_t)B_ * H_, woutb, b_out, nullptr, out, 256, H_);
}
